// decay_Attention_33689723470286
// MI455X (gfx1250) — matrix operand order repaired and checked by reading, not compiled, not run on hardware
//
#include <hip/hip_runtime.h>
#include <hip/hip_bf16.h>
#include <stdint.h>

// ---------------- problem constants ----------------
#define D_MODEL 1024
#define N_HEADS 16
#define D_K     64
#define BATCH   2048
#define SEQ     20
#define M_ROWS  (BATCH * SEQ)          // 40960
#define QN      (M_ROWS * D_MODEL)     // 41943040 floats in Q
#define WN      (D_MODEL * D_MODEL)    // 1048576 per weight matrix
#define QKVN    (BATCH * N_HEADS * SEQ * D_K) // 41943040 per projection

// workspace layout (in ushort/bf16 units)
#define WBF_OFF 0
#define QBF_OFF (3 * WN)               // 3145728
#define QKV_OFF (QBF_OFF + QN)         // 45088768
// total ws: (QKV_OFF + 3*QKVN) * 2 bytes ~= 326 MiB

typedef __attribute__((ext_vector_type(16))) __bf16 v16bf;
typedef __attribute__((ext_vector_type(8)))  float  v8f;

// branchless fp32 -> bf16 round-to-nearest-even (all values here are finite)
static __device__ __forceinline__ unsigned short f2bf(float f) {
    union { float f; uint32_t u; } c; c.f = f;
    uint32_t u = c.u;
    return (unsigned short)((u + 0x7fffu + ((u >> 16) & 1u)) >> 16);
}

static __device__ __forceinline__ v16bf mkfrag(uint4 a, uint4 b) {
    union { uint4 u[2]; v16bf v; } x;
    x.u[0] = a; x.u[1] = b;
    return x.v;
}

// async global->LDS copy of 32 bytes (two b128), same +16B delta on both sides
static __device__ __forceinline__ void async_cp32(const unsigned short* g, unsigned short* l) {
    uint32_t lds = (uint32_t)(size_t)l;   // low 32 bits of generic LDS addr = LDS byte address
    asm volatile("global_load_async_to_lds_b128 %0, %1, off\n\t"
                 "global_load_async_to_lds_b128 %0, %1, off offset:16"
                 :: "v"(lds), "v"(g) : "memory");
}

// ---------------- kernel A: fp32 -> bf16 conversion of Q and W_{Q,K,V} ----------------
__global__ __launch_bounds__(256) void cvt_kernel(const float* __restrict__ Q,
                                                  const float* __restrict__ WQ,
                                                  const float* __restrict__ WK,
                                                  const float* __restrict__ WV,
                                                  unsigned short* __restrict__ ws) {
    int i = blockIdx.x * 256 + threadIdx.x;
    int base = i * 4;
    if (base >= QN + 3 * WN) return;
    const float* src;
    unsigned short* dst;
    if (base < QN) {
        src = Q + base;
        dst = ws + QBF_OFF + base;
    } else {
        int w = base - QN;
        int p = w >> 20;              // WN == 2^20
        int off = w & (WN - 1);
        src = (p == 0 ? WQ : (p == 1 ? WK : WV)) + off;
        dst = ws + WBF_OFF + p * WN + off;
    }
    float4 v = *(const float4*)src;
    ushort4 o;
    o.x = f2bf(v.x); o.y = f2bf(v.y); o.z = f2bf(v.z); o.w = f2bf(v.w);
    *(ushort4*)dst = o;
}

// ---------------- kernel B: projection GEMM  y = x @ W^T + b (bf16 WMMA) ----------------
// block tile 128x128, 8 waves, each wave 64x32 (4x2 WMMA tiles), K-step 32,
// double-buffered async global->LDS staging, LDS-bounced vectorized epilogue
#define LSTR 40    // LDS row stride (bf16) for A/B staging
#define CSTR 136   // LDS row stride (bf16) for C tile bounce

__global__ __launch_bounds__(256) void proj_kernel(unsigned short* __restrict__ ws,
                                                   const float* __restrict__ bQ,
                                                   const float* __restrict__ bK,
                                                   const float* __restrict__ bV) {
    __shared__ unsigned short As[2][128 * LSTR];
    __shared__ unsigned short Bs[2][128 * LSTR];
    __shared__ unsigned short Ct[128 * CSTR];

    const int p = blockIdx.y;                      // 0:Q 1:K 2:V
    const unsigned short* Qbf = ws + QBF_OFF;
    const unsigned short* Wbf = ws + WBF_OFF + p * WN;
    unsigned short* Out = ws + QKV_OFF + p * QKVN;
    const float* bias = (p == 0) ? bQ : ((p == 1) ? bK : bV);

    const int tid = threadIdx.x;
    const int lane = tid & 31;
    const int wv = tid >> 5;            // 0..7
    const int hf = lane >> 4;           // lane half
    const int l16 = lane & 15;
    const int wm = wv & 1;              // wave row  (64 rows)
    const int wn = wv >> 1;             // wave col  (32 cols)

    const int m0 = (blockIdx.x >> 3) * 128;
    const int n0 = (blockIdx.x & 7) * 128;

    const v8f vzero = {0.f, 0.f, 0.f, 0.f, 0.f, 0.f, 0.f, 0.f};
    v8f acc[4][2];
#pragma unroll
    for (int i = 0; i < 4; ++i)
#pragma unroll
        for (int j = 0; j < 2; ++j) acc[i][j] = vzero;

    const int srow = tid >> 1;          // staging row 0..127
    const int scol = (tid & 1) * 16;    // staging col 0 or 16

    const unsigned short* gA = Qbf + (m0 + srow) * D_MODEL + scol;
    const unsigned short* gB = Wbf + (n0 + srow) * D_MODEL + scol;
    unsigned short* lA[2] = { &As[0][srow * LSTR + scol], &As[1][srow * LSTR + scol] };
    unsigned short* lB[2] = { &Bs[0][srow * LSTR + scol], &Bs[1][srow * LSTR + scol] };

    // prologue: fill buffer 0
    async_cp32(gA, lA[0]);
    async_cp32(gB, lB[0]);
    asm volatile("s_wait_asynccnt 0" ::: "memory");
    __syncthreads();

    for (int kk = 0; kk < D_MODEL; kk += 32) {
        const int cur = (kk >> 5) & 1;
        // kick off DMA for the next K-step into the other buffer (overlaps WMMAs)
        if (kk + 32 < D_MODEL) {
            async_cp32(gA + kk + 32, lA[cur ^ 1]);
            async_cp32(gB + kk + 32, lB[cur ^ 1]);
            __builtin_prefetch(gA + kk + 64, 0, 1);
            __builtin_prefetch(gB + kk + 64, 0, 1);
        }

        v16bf bfr[2];
#pragma unroll
        for (int ni = 0; ni < 2; ++ni) {
            int n = wn * 32 + ni * 16 + l16;
            int base = n * LSTR + hf * 16;
            bfr[ni] = mkfrag(*(const uint4*)&Bs[cur][base], *(const uint4*)&Bs[cur][base + 8]);
        }
#pragma unroll
        for (int mi = 0; mi < 4; ++mi) {
            int row = wm * 64 + mi * 16 + l16;
            int base = row * LSTR;
            v16bf afr = mkfrag(*(const uint4*)&As[cur][base + hf * 16],
                               *(const uint4*)&As[cur][base + hf * 16 + 8]);
#pragma unroll
            for (int ni = 0; ni < 2; ++ni) {
                acc[mi][ni] = __builtin_amdgcn_wmma_f32_16x16x32_bf16(
                    false, afr, false, bfr[ni], (short)0, acc[mi][ni], false, false);
            }
        }

        asm volatile("s_wait_asynccnt 0" ::: "memory");
        __syncthreads();   // next buffer filled, current buffer fully consumed
    }

    // epilogue 1: +bias, dump C tile (bf16) to LDS in (row, col) layout
#pragma unroll
    for (int mi = 0; mi < 4; ++mi) {
#pragma unroll
        for (int ni = 0; ni < 2; ++ni) {
            int col = wn * 32 + ni * 16 + l16;
            float bv = bias[n0 + col];
#pragma unroll
            for (int r = 0; r < 8; ++r) {
                int row = wm * 64 + mi * 16 + r + hf * 8;
                Ct[row * CSTR + col] = f2bf(acc[mi][ni][r] + bv);
            }
        }
    }
    __syncthreads();

    // epilogue 2: cooperative contiguous store into [b,h,s,d] layout,
    // one (row, head) 64-element chunk per thread -> 8x global_store_b128
    {
        const int lr = tid >> 1;             // local row 0..127
        const int nh = tid & 1;              // head half within the 128-col tile
        const int m = m0 + lr;
        const int bb = m / SEQ;
        const int s = m - bb * SEQ;
        const int hh = (n0 >> 6) + nh;
        const uint4* sp = (const uint4*)&Ct[lr * CSTR + nh * 64];
        uint4* dp = (uint4*)(Out + ((bb * N_HEADS + hh) * SEQ + s) * D_K);
#pragma unroll
        for (int q = 0; q < 8; ++q) dp[q] = sp[q];
    }
}

// ---------------- kernel C: attention, one wave per (b,h) ----------------
__global__ __launch_bounds__(128) void attn_kernel(const unsigned short* __restrict__ ws,
                                                   const int* __restrict__ length,
                                                   float* __restrict__ out) {
    __shared__ unsigned short sP[4][32 * LSTR];   // P = attn + Md, bf16, per wave
    __shared__ unsigned short sV[4][64 * LSTR];   // v transposed [d][s], bf16, per wave

    const int tid = threadIdx.x;
    const int lane = tid & 31;
    const int wv = tid >> 5;
    const int hf = lane >> 4;
    const int l16 = lane & 15;

    const int pair = blockIdx.x * 4 + wv;         // (b,h)
    const int b = pair >> 4;
    const int h = pair & 15;

    const unsigned short* qkv = ws + QKV_OFF;
    const int toff = (b * N_HEADS + h) * (SEQ * D_K);
    const unsigned short* qp = qkv + toff;
    const unsigned short* kp = qkv + QKVN + toff;
    const unsigned short* vp = qkv + 2 * QKVN + toff;
    const int len = length[b];

    // ---- load q (A-layout) and k (B-layout of k^T) fragments from global, clamp pad rows
    v16bf qf[2][2], kf[2][2];
#pragma unroll
    for (int mt = 0; mt < 2; ++mt)
#pragma unroll
        for (int kt = 0; kt < 2; ++kt) {
            int row = mt * 16 + l16; if (row > SEQ - 1) row = SEQ - 1;
            int off = row * D_K + kt * 32;
            qf[mt][kt] = mkfrag(*(const uint4*)(qp + off + hf * 16),
                                *(const uint4*)(qp + off + hf * 16 + 8));
        }
#pragma unroll
    for (int nt = 0; nt < 2; ++nt)
#pragma unroll
        for (int kt = 0; kt < 2; ++kt) {
            int row = nt * 16 + l16; if (row > SEQ - 1) row = SEQ - 1;
            int off = row * D_K + kt * 32 + hf * 16;
            kf[nt][kt] = mkfrag(*(const uint4*)(kp + off),
                                *(const uint4*)(kp + off + 8));
        }

    // ---- stage v transposed into LDS (zero-padded to 32 rows)
    unsigned short* vT = sV[wv];
    for (int idx = lane; idx < 64 * 32; idx += 32) {
        int d = idx >> 5, s = idx & 31;
        vT[d * LSTR + s] = 0;
    }
    for (int idx = lane; idx < SEQ * D_K; idx += 32) {
        int s = idx >> 6, d = idx & 63;
        vT[d * LSTR + s] = vp[idx];
    }

    // ---- scores = q k^T  (8 WMMAs)
    const v8f vzero = {0.f, 0.f, 0.f, 0.f, 0.f, 0.f, 0.f, 0.f};
    v8f sacc[2][2];
#pragma unroll
    for (int mt = 0; mt < 2; ++mt)
#pragma unroll
        for (int nt = 0; nt < 2; ++nt) {
            v8f c = vzero;
#pragma unroll
            for (int kt = 0; kt < 2; ++kt)
                c = __builtin_amdgcn_wmma_f32_16x16x32_bf16(
                        false, qf[mt][kt], false, kf[nt][kt], (short)0, c, false, false);
            sacc[mt][nt] = c;
        }

    // ---- exp(scale*s) * keymask, row sums, attn, P = attn + Md  -> LDS (bf16)
    float ev[2][2][8];
#pragma unroll
    for (int mt = 0; mt < 2; ++mt)
#pragma unroll
        for (int nt = 0; nt < 2; ++nt)
#pragma unroll
            for (int r = 0; r < 8; ++r) {
                int j = nt * 16 + l16;
                float sc = sacc[mt][nt][r] * 0.125f;   // 1/sqrt(64)
                ev[mt][nt][r] = (j < len && j < SEQ) ? __expf(sc) : 0.0f;
            }

    float rs[2][8];
#pragma unroll
    for (int mt = 0; mt < 2; ++mt)
#pragma unroll
        for (int r = 0; r < 8; ++r) {
            float s = ev[mt][0][r] + ev[mt][1][r];
#pragma unroll
            for (int x = 1; x < 16; x <<= 1) s += __shfl_xor(s, x, 32); // stays in 16-lane half
            rs[mt][r] = s;
        }

    unsigned short* pl = sP[wv];
#pragma unroll
    for (int mt = 0; mt < 2; ++mt)
#pragma unroll
        for (int nt = 0; nt < 2; ++nt)
#pragma unroll
            for (int r = 0; r < 8; ++r) {
                int i = mt * 16 + r + hf * 8;
                int j = nt * 16 + l16;
                float attn = ev[mt][nt][r] / (rs[mt][r] + 1e-8f);
                int dji = i - j; if (dji < 0) dji = -dji;
                float md = (j < SEQ) ? -(float)dji : 0.0f;   // Md NOT length-masked
                pl[i * LSTR + j] = f2bf(attn + md);
            }

    __syncthreads();

    // ---- out = P @ v  (8 WMMAs), K = 32 (padded S)
    v16bf pf[2];
#pragma unroll
    for (int mt = 0; mt < 2; ++mt) {
        int i = mt * 16 + l16;
        int base = i * LSTR;
        pf[mt] = mkfrag(*(const uint4*)&pl[base + hf * 16],
                        *(const uint4*)&pl[base + hf * 16 + 8]);
    }
    v16bf vf[4];
#pragma unroll
    for (int nt = 0; nt < 4; ++nt) {
        int d = nt * 16 + l16;
        int base = d * LSTR + hf * 16;
        vf[nt] = mkfrag(*(const uint4*)&vT[base], *(const uint4*)&vT[base + 8]);
    }

#pragma unroll
    for (int mt = 0; mt < 2; ++mt)
#pragma unroll
        for (int nt = 0; nt < 4; ++nt) {
            v8f o = __builtin_amdgcn_wmma_f32_16x16x32_bf16(
                        false, pf[mt], false, vf[nt], (short)0, vzero, false, false);
#pragma unroll
            for (int r = 0; r < 8; ++r) {
                int s = mt * 16 + r + hf * 8;
                if (s < SEQ) {
                    int col = h * D_K + nt * 16 + l16;
                    out[(b * SEQ + s) * D_MODEL + col] = o[r];
                }
            }
        }
}

// ---------------- launcher ----------------
extern "C" void kernel_launch(void* const* d_in, const int* in_sizes, int n_in,
                              void* d_out, int out_size, void* d_ws, size_t ws_size,
                              hipStream_t stream) {
    const float* Q  = (const float*)d_in[0];
    const float* WQ = (const float*)d_in[1];
    const float* bQ = (const float*)d_in[2];
    const float* WK = (const float*)d_in[3];
    const float* bK = (const float*)d_in[4];
    const float* WV = (const float*)d_in[5];
    const float* bV = (const float*)d_in[6];
    const int* length = (const int*)d_in[7];
    unsigned short* ws = (unsigned short*)d_ws;
    float* out = (float*)d_out;

    // A: convert Q + 3 W's to bf16
    {
        int total4 = (QN + 3 * WN) / 4;
        int blocks = (total4 + 255) / 256;
        cvt_kernel<<<blocks, 256, 0, stream>>>(Q, WQ, WK, WV, ws);
    }
    // B: three projections, 128x128 block tiles
    {
        dim3 grid((M_ROWS / 128) * (D_MODEL / 128), 3);
        proj_kernel<<<grid, 256, 0, stream>>>(ws, bQ, bK, bV);
    }
    // C: attention + decay, one wave per (b,h)
    {
        attn_kernel<<<(BATCH * N_HEADS) / 4, 128, 0, stream>>>(ws, length, out);
    }
}
